// FluxAttention_35768487641494
// MI455X (gfx1250) — matrix operand order repaired and checked by reading, not compiled, not run on hardware
//
#include <hip/hip_runtime.h>
#include <hip/hip_bf16.h>

typedef __attribute__((ext_vector_type(16))) __bf16        v16bf;
typedef __attribute__((ext_vector_type(8)))  float         v8f;
typedef __attribute__((ext_vector_type(8)))  unsigned int  v8u;

#define S_TOT   2304
#define S_TXT   256
#define S_IMG   2048
#define DMODEL  3072
#define HEADS   24
#define HD      128

// ---------- helpers ----------

__device__ __forceinline__ unsigned short f2bf(float f) {
  unsigned int u = __builtin_bit_cast(unsigned int, f);
  u += 0x7fffu + ((u >> 16) & 1u);          // round-to-nearest-even
  return (unsigned short)(u >> 16);
}

// A-fragment (16x32 bf16): lane m = l&15; K chunks [c0..c0+7] and [c0+16..c0+23], c0 = grp*8.
// base must point at element c0 of the row; reads 8 + 8 bf16 with a 16-element gap.
__device__ __forceinline__ v16bf frag_split(const unsigned short* base) {
  const unsigned int* p0 = (const unsigned int*)base;
  const unsigned int* p1 = (const unsigned int*)(base + 16);
  v8u u;
  u[0] = p0[0]; u[1] = p0[1]; u[2] = p0[2]; u[3] = p0[3];
  u[4] = p1[0]; u[5] = p1[1]; u[6] = p1[2]; u[7] = p1[3];
  return __builtin_bit_cast(v16bf, u);
}

// B-fragment (32x16 bf16): lane n = l&15; K = grp*16 + 0..15 contiguous.
__device__ __forceinline__ v16bf frag_contig(const unsigned short* base) {
  const unsigned int* p = (const unsigned int*)base;
  v8u u;
#pragma unroll
  for (int i = 0; i < 8; ++i) u[i] = p[i];
  return __builtin_bit_cast(v16bf, u);
}

__device__ __forceinline__ v8f wmma_bf16(v16bf a, v16bf b, v8f c) {
  return __builtin_amdgcn_wmma_f32_16x16x32_bf16(false, a, false, b, (short)0, c,
                                                 false, false);
}

// ---------- GEMM: C[m,n] = A[m,:] @ W[:,n] (+ bias), bf16 WMMA, fp32 accumulate ----------
// Block tile 128x128, BK=32, 256 threads = 8 waves in a 4(M) x 2(N) grid,
// each wave owns a 32x64 subtile = 2x4 WMMA accumulators.
// Rows [0,256) use Aenc/Wenc/BiasEnc, rows [256,2304) use Aimg/Wimg/BiasImg.
// mode 0: fp32 out [M][N];  mode 1: bf16 out [h][s][128];  mode 2: split fp32 out.
__global__ void __launch_bounds__(256)
gemm_bf16(const float* __restrict__ Aenc, const float* __restrict__ Aimg,
          const float* __restrict__ Wenc, const float* __restrict__ Wimg,
          const float* __restrict__ BiasEnc, const float* __restrict__ BiasImg,
          float* __restrict__ OutF, unsigned short* __restrict__ OutBf,
          float* __restrict__ OutF2, int mode)
{
  const int tid  = threadIdx.x;
  const int lane = tid & 31;
  const int ln   = lane & 15;
  const int grp  = lane >> 4;
  const int w    = tid >> 5;
  const int wm   = w & 3;        // 32-row subtile
  const int wn   = w >> 2;       // 64-col subtile
  const int n0   = blockIdx.x * 128;
  const int m0   = blockIdx.y * 128;
  const bool enc = (m0 < S_TXT);
  const float* A    = enc ? (Aenc + (size_t)m0 * DMODEL)
                          : (Aimg + (size_t)(m0 - S_TXT) * DMODEL);
  const float* W    = enc ? Wenc : Wimg;
  const float* bias = enc ? BiasEnc : BiasImg;

  __shared__ unsigned short As[128][34];   // [m][k]  bf16
  __shared__ unsigned short Bt[128][34];   // [n][k]  bf16 (W tile transposed)

  const v8f vz = {0.f, 0.f, 0.f, 0.f, 0.f, 0.f, 0.f, 0.f};
  v8f acc[2][4];
#pragma unroll
  for (int i = 0; i < 2; ++i)
#pragma unroll
    for (int j = 0; j < 4; ++j) acc[i][j] = vz;

  for (int kt = 0; kt < DMODEL; kt += 32) {
    __syncthreads();
    // A tile: 128x32 fp32 -> bf16 LDS
#pragma unroll
    for (int it = 0; it < 4; ++it) {
      int idx = (it * 256 + tid) * 4;
      int r = idx >> 5, c = idx & 31;
      float4 f = *(const float4*)(A + (size_t)r * DMODEL + kt + c);
      As[r][c + 0] = f2bf(f.x);
      As[r][c + 1] = f2bf(f.y);
      As[r][c + 2] = f2bf(f.z);
      As[r][c + 3] = f2bf(f.w);
    }
    // W tile: 32x128 fp32 -> transposed bf16 LDS
#pragma unroll
    for (int it = 0; it < 4; ++it) {
      int idx = (it * 256 + tid) * 4;
      int k = idx >> 7, n = idx & 127;
      float4 f = *(const float4*)(W + (size_t)(kt + k) * DMODEL + n0 + n);
      Bt[n + 0][k] = f2bf(f.x);
      Bt[n + 1][k] = f2bf(f.y);
      Bt[n + 2][k] = f2bf(f.z);
      Bt[n + 3][k] = f2bf(f.w);
    }
    if (kt + 32 < DMODEL) {   // lowers to global_prefetch_b8
      __builtin_prefetch(A + (size_t)(tid >> 1) * DMODEL + kt + 32, 0, 1);
      __builtin_prefetch(W + (size_t)(kt + 32 + (tid & 31)) * DMODEL + n0, 0, 1);
    }
    __syncthreads();

    v16bf af[2], bfr[4];
#pragma unroll
    for (int i = 0; i < 2; ++i)
      af[i] = frag_contig(&As[wm * 32 + i * 16 + ln][grp * 16]);
#pragma unroll
    for (int j = 0; j < 4; ++j)
      bfr[j] = frag_contig(&Bt[wn * 64 + j * 16 + ln][grp * 16]);
#pragma unroll
    for (int i = 0; i < 2; ++i)
#pragma unroll
      for (int j = 0; j < 4; ++j)
        acc[i][j] = wmma_bf16(af[i], bfr[j], acc[i][j]);
  }

  // epilogue: C/D layout -> lane holds col n = ln (+16*grp group rows), VGPR v -> row grp*8+v
#pragma unroll
  for (int i = 0; i < 2; ++i) {
#pragma unroll
    for (int j = 0; j < 4; ++j) {
      const int ncol  = n0 + wn * 64 + j * 16 + ln;
      const float bv  = bias ? bias[ncol] : 0.0f;
      const int mbase = m0 + wm * 32 + i * 16 + grp * 8;
#pragma unroll
      for (int v = 0; v < 8; ++v) {
        const int m     = mbase + v;
        const float val = acc[i][j][v] + bv;
        if (mode == 0) {
          OutF[(size_t)m * DMODEL + ncol] = val;
        } else if (mode == 1) {
          const int h = ncol >> 7, d = ncol & 127;
          OutBf[((size_t)h * S_TOT + m) * HD + d] = f2bf(val);
        } else {
          if (enc) OutF [(size_t)m * DMODEL + ncol] = val;
          else     OutF2[(size_t)(m - S_TXT) * DMODEL + ncol] = val;
        }
      }
    }
  }
}

// ---------- RMSNorm (head_dim=128, fp32 stats) + interleaved-pair RoPE -> bf16 [h][s][128] ----------
__global__ void __launch_bounds__(256)
rms_rope(const float* __restrict__ raw, const float* __restrict__ cosb,
         const float* __restrict__ sinb, const float* __restrict__ scMain,
         const float* __restrict__ scAdd, unsigned short* __restrict__ outb)
{
  const int r    = blockIdx.x * 8 + (threadIdx.x >> 5);  // (s,h) row
  const int lane = threadIdx.x & 31;
  const int s    = r / HEADS;
  const int h    = r % HEADS;
  const int d0   = lane * 4;
  const float4 x = *(const float4*)(raw + (size_t)s * DMODEL + h * HD + d0);
  float ss = x.x * x.x + x.y * x.y + x.z * x.z + x.w * x.w;
#pragma unroll
  for (int m = 1; m < 32; m <<= 1) ss += __shfl_xor(ss, m, 32);
  const float inv = rsqrtf(ss * (1.0f / 128.0f) + 1e-6f);
  const float* sc = (s < S_TXT) ? scAdd : scMain;
  const float y0 = x.x * inv * sc[d0 + 0];
  const float y1 = x.y * inv * sc[d0 + 1];
  const float y2 = x.z * inv * sc[d0 + 2];
  const float y3 = x.w * inv * sc[d0 + 3];
  const float4 c  = *(const float4*)(cosb + (size_t)s * HD + d0);
  const float4 sn = *(const float4*)(sinb + (size_t)s * HD + d0);
  unsigned short* dst = outb + ((size_t)h * S_TOT + s) * HD + d0;
  dst[0] = f2bf(y0 * c.x - y1 * sn.x);
  dst[1] = f2bf(y1 * c.y + y0 * sn.y);
  dst[2] = f2bf(y2 * c.z - y3 * sn.z);
  dst[3] = f2bf(y3 * c.w + y2 * sn.w);
}

// ---------- flash attention: one (head, 64 q-rows) per block; 4 waves x 16 rows ----------
__global__ void __launch_bounds__(128)
attn_fwd(const unsigned short* __restrict__ Qb,
         const unsigned short* __restrict__ Kb,
         const unsigned short* __restrict__ Vb,
         float* __restrict__ O)
{
  const int tid  = threadIdx.x;
  const int lane = tid & 31;
  const int ln   = lane & 15;
  const int grp  = lane >> 4;
  const int w    = tid >> 5;
  const int h    = blockIdx.y;
  const int q0   = blockIdx.x * 64;
  const size_t hbase = (size_t)h * S_TOT * HD;

  __shared__ unsigned short Vt[128][66];      // V tile transposed: [d][kk]
  __shared__ unsigned short Pl[4][16][64];    // per-wave P for C->A layout swap

  // Q A-fragments (row = lane, contiguous head-dim) held in registers for all K tiles
  v16bf aq[4];
  {
    const unsigned short* qrow = Qb + hbase + (size_t)(q0 + w * 16 + ln) * HD;
#pragma unroll
    for (int ks = 0; ks < 4; ++ks)
      aq[ks] = frag_contig(qrow + ks * 32 + grp * 16);
  }

  const v8f vz = {0.f, 0.f, 0.f, 0.f, 0.f, 0.f, 0.f, 0.f};
  v8f oacc[8];
#pragma unroll
  for (int jo = 0; jo < 8; ++jo) oacc[jo] = vz;
  float mrow[8], lrow[8];
#pragma unroll
  for (int v = 0; v < 8; ++v) { mrow[v] = -__builtin_inff(); lrow[v] = 0.f; }

  const float sm = 0.08838834764831845f;      // 1/sqrt(128)

  for (int j0 = 0; j0 < S_TOT; j0 += 64) {
    __syncthreads();
    // V tile 64x128 -> Vt[d][kk]
#pragma unroll
    for (int it = 0; it < 16; ++it) {
      int idx = it * 512 + tid * 4;
      int kk = idx >> 7, d = idx & 127;
      const unsigned short* src = Vb + hbase + (size_t)(j0 + kk) * HD + d;
      uint2 f = *(const uint2*)src;
      Vt[d + 0][kk] = (unsigned short)(f.x & 0xffffu);
      Vt[d + 1][kk] = (unsigned short)(f.x >> 16);
      Vt[d + 2][kk] = (unsigned short)(f.y & 0xffffu);
      Vt[d + 3][kk] = (unsigned short)(f.y >> 16);
    }
    __syncthreads();

    // S = Q @ K^T  (K rows are contiguous in head-dim -> direct global B-fragments)
    v8f sacc[4];
#pragma unroll
    for (int j = 0; j < 4; ++j) sacc[j] = vz;
#pragma unroll
    for (int ks = 0; ks < 4; ++ks) {
#pragma unroll
      for (int j = 0; j < 4; ++j) {
        const unsigned short* krow =
            Kb + hbase + (size_t)(j0 + j * 16 + ln) * HD + ks * 32 + grp * 16;
        sacc[j] = wmma_bf16(aq[ks], frag_contig(krow), sacc[j]);
      }
    }
#pragma unroll
    for (int j = 0; j < 4; ++j)
#pragma unroll
      for (int v = 0; v < 8; ++v) sacc[j][v] *= sm;

    // online softmax (row = grp*8+v lives in VGPR v across the 16-lane half-wave)
    float rsum[8];
#pragma unroll
    for (int v = 0; v < 8; ++v) {
      float x = sacc[0][v];
#pragma unroll
      for (int j = 1; j < 4; ++j) x = fmaxf(x, sacc[j][v]);
#pragma unroll
      for (int msk = 1; msk < 16; msk <<= 1) x = fmaxf(x, __shfl_xor(x, msk, 32));
      const float mnew  = fmaxf(mrow[v], x);
      const float alpha = __expf(mrow[v] - mnew);
      mrow[v] = mnew;
      lrow[v] *= alpha;
#pragma unroll
      for (int jo = 0; jo < 8; ++jo) oacc[jo][v] *= alpha;
      float s = 0.f;
#pragma unroll
      for (int j = 0; j < 4; ++j) {
        const float p = __expf(sacc[j][v] - mnew);
        s += p;
        Pl[w][grp * 8 + v][j * 16 + ln] = f2bf(p);
      }
      rsum[v] = s;
    }
#pragma unroll
    for (int v = 0; v < 8; ++v) {
      float s = rsum[v];
#pragma unroll
      for (int msk = 1; msk < 16; msk <<= 1) s += __shfl_xor(s, msk, 32);
      lrow[v] += s;
    }

    // O += P @ V  (P re-read from LDS in A-layout, V from transposed LDS tile)
#pragma unroll
    for (int ks = 0; ks < 2; ++ks) {
      v16bf ap = frag_contig(&Pl[w][ln][ks * 32 + grp * 16]);
#pragma unroll
      for (int jo = 0; jo < 8; ++jo) {
        v16bf bv = frag_contig(&Vt[jo * 16 + ln][ks * 32 + grp * 16]);
        oacc[jo] = wmma_bf16(ap, bv, oacc[jo]);
      }
    }
  }

  // normalize and write O fp32 [s][h*128+d]
#pragma unroll
  for (int jo = 0; jo < 8; ++jo) {
#pragma unroll
    for (int v = 0; v < 8; ++v) {
      const int m = q0 + w * 16 + grp * 8 + v;
      O[(size_t)m * DMODEL + h * HD + jo * 16 + ln] = oacc[jo][v] / lrow[v];
    }
  }
}

// ---------- launch ----------

extern "C" void kernel_launch(void* const* d_in, const int* in_sizes, int n_in,
                              void* d_out, int out_size, void* d_ws, size_t ws_size,
                              hipStream_t stream) {
  (void)in_sizes; (void)n_in; (void)out_size; (void)ws_size;
  const float* hidden     = (const float*)d_in[0];
  const float* encoder    = (const float*)d_in[1];
  const float* cosb       = (const float*)d_in[2];
  const float* sinb       = (const float*)d_in[3];
  const float* wq         = (const float*)d_in[4];
  const float* wk         = (const float*)d_in[5];
  const float* wv         = (const float*)d_in[6];
  const float* wq_add     = (const float*)d_in[7];
  const float* wk_add     = (const float*)d_in[8];
  const float* wv_add     = (const float*)d_in[9];
  const float* bq_add     = (const float*)d_in[10];
  const float* bk_add     = (const float*)d_in[11];
  const float* bv_add     = (const float*)d_in[12];
  const float* norm_q     = (const float*)d_in[13];
  const float* norm_k     = (const float*)d_in[14];
  const float* norm_add_q = (const float*)d_in[15];
  const float* norm_add_k = (const float*)d_in[16];
  const float* w_out      = (const float*)d_in[17];
  const float* b_out      = (const float*)d_in[18];
  const float* w_add_out  = (const float*)d_in[19];
  const float* b_add_out  = (const float*)d_in[20];

  char* ws = (char*)d_ws;
  const size_t SZF = (size_t)S_TOT * DMODEL * sizeof(float);
  const size_t SZB = (size_t)S_TOT * DMODEL * sizeof(unsigned short);
  float*          Qraw  = (float*)(ws);
  float*          Kraw  = (float*)(ws + SZF);
  unsigned short* Qb    = (unsigned short*)(ws + 2 * SZF);
  unsigned short* Kb    = (unsigned short*)(ws + 2 * SZF + SZB);
  unsigned short* Vb    = (unsigned short*)(ws + 2 * SZF + 2 * SZB);
  float*          Oattn = (float*)(ws + 2 * SZF + 3 * SZB);

  float* out_img = (float*)d_out;
  float* out_enc = (float*)d_out + (size_t)S_IMG * DMODEL;

  dim3 gg(DMODEL / 128, S_TOT / 128);
  gemm_bf16<<<gg, 256, 0, stream>>>(encoder, hidden, wq_add, wq, bq_add, nullptr,
                                    Qraw, nullptr, nullptr, 0);
  gemm_bf16<<<gg, 256, 0, stream>>>(encoder, hidden, wk_add, wk, bk_add, nullptr,
                                    Kraw, nullptr, nullptr, 0);
  gemm_bf16<<<gg, 256, 0, stream>>>(encoder, hidden, wv_add, wv, bv_add, nullptr,
                                    nullptr, Vb, nullptr, 1);
  rms_rope<<<(S_TOT * HEADS) / 8, 256, 0, stream>>>(Qraw, cosb, sinb, norm_q,
                                                    norm_add_q, Qb);
  rms_rope<<<(S_TOT * HEADS) / 8, 256, 0, stream>>>(Kraw, cosb, sinb, norm_k,
                                                    norm_add_k, Kb);
  attn_fwd<<<dim3(S_TOT / 64, HEADS), 128, 0, stream>>>(Qb, Kb, Vb, Oattn);
  gemm_bf16<<<gg, 256, 0, stream>>>(Oattn, Oattn + (size_t)S_TXT * DMODEL,
                                    w_add_out, w_out, b_add_out, b_out,
                                    out_enc, nullptr, out_img, 2);
}
